// classifier_78563541778828
// MI455X (gfx1250) — matrix operand order repaired and checked by reading, not compiled, not run on hardware
//
#include <hip/hip_runtime.h>
#include <hip/hip_bf16.h>
#include <math.h>

// ---------------- problem constants ----------------
#define BD   256      // B
#define LD   6        // L
#define ND   16       // N (graph nodes)
#define DD   768      // D
#define MD   4        // M = M1+M2
#define M1D  2
#define HD   2        // heads in GAT
#define OD   32
#define NHD  12
#define DHD  64       // D/NH
#define FFD  3072
#define HIDD 128
#define NLB  10
#define BLD  (BD*LD)          // 1536
#define TOK  (BD*LD*ND)       // 24576

typedef __bf16 bf16_t;
typedef __attribute__((ext_vector_type(16))) __bf16 v16bf;
typedef __attribute__((ext_vector_type(8)))  float  v8f;
typedef __attribute__((ext_vector_type(4)))  unsigned int u32x4;
typedef __attribute__((ext_vector_type(8)))  int          i32x8;
typedef __attribute__((ext_vector_type(4)))  int          i32x4;

union FragU { v16bf v; unsigned int u[8]; };

// ---- WMMA fragment loaders (contiguous k order k = 16h + s, same for A and B) ----
// A (16x32 bf16, row-major, lda in elements, lda must be even):
//   row = lane&15, kbase = 16*(lane>=16); VGPR i holds k = kbase + 2i
__device__ inline v16bf load_frag_a(const bf16_t* A, int lda, int lane, int k0) {
  int r  = lane & 15;
  int kb = (lane >> 4) << 4;
  const unsigned int* p = (const unsigned int*)(A + (size_t)r * lda + k0 + kb);
  FragU f;
#pragma unroll
  for (int i = 0; i < 4; ++i) f.u[i] = p[i];
#pragma unroll
  for (int i = 4; i < 8; ++i) f.u[i] = p[i];
  return f.v;
}

// B (32x16 bf16) from K-major ("transposed") weights Wt[n][k], ldb in elements:
//   col = lane&15, k = 16*(lane>=16) + 2i  -> 8 contiguous u32 per lane
__device__ inline v16bf load_frag_b(const bf16_t* Bt, int ldb, int lane, int k0) {
  int c  = lane & 15;
  int kb = (lane >> 4) << 4;
  const unsigned int* p = (const unsigned int*)(Bt + (size_t)c * ldb + k0 + kb);
  FragU f;
#pragma unroll
  for (int i = 0; i < 8; ++i) f.u[i] = p[i];
  return f.v;
}

__device__ inline v8f wmma_bf16(v16bf a, v16bf b, v8f c) {
  return __builtin_amdgcn_wmma_f32_16x16x32_bf16(false, a, false, b, (short)0, c,
                                                 false, false);
}

__device__ inline float apply_act(float v, int act) {
  if (act == 1) return v >= 0.f ? v : 0.1f * v;                 // lrelu 0.1
  if (act == 2) {                                               // gelu (tanh approx)
    float c = v + 0.044715f * v * v * v;
    return 0.5f * v * (1.f + tanhf(0.7978845608028654f * c));
  }
  if (act == 3) return tanhf(v);
  return v;
}

// ---------------- weight transpose fp32 -> bf16 K-major ----------------
// W [batch][K][N] row-major  ->  Wt [batch][N][K]
__global__ void transpose_to_bf16(const float* __restrict__ W, bf16_t* __restrict__ Wt,
                                  int K, int Nc, int total) {
  int idx = blockIdx.x * blockDim.x + threadIdx.x;
  if (idx >= total) return;
  int b = idx / (K * Nc);
  int r = idx - b * K * Nc;
  int k = r / Nc, n = r - k * Nc;
  Wt[(size_t)b * K * Nc + (size_t)n * K + k] = (bf16_t)W[idx];
}

// ---------------- embedding gather ----------------
__global__ void embed_kernel(const int* __restrict__ wid, const int* __restrict__ pid,
                             const float* __restrict__ we, const float* __restrict__ pe,
                             bf16_t* __restrict__ hb) {
  int row = blockIdx.x;
  int w = wid[row], p = pid[row];
  const float* wr = we + (size_t)w * DD;
  const float* pr = pe + (size_t)p * DD;
  bf16_t* o = hb + (size_t)row * DD;
#pragma unroll
  for (int j = 0; j < 3; ++j) {
    int d = threadIdx.x * 3 + j;
    o[d] = (bf16_t)(wr[d] + pr[d]);
  }
}

// ---------------- GAT block: one workgroup per (b,l,m) ----------------
// A-tile (16x768 bf16 = 24 KB) staged into LDS once via the Tensor Data Mover
// (TENSOR_LOAD_TO_LDS, TENSORcnt), then read by all 4 waves as WMMA A-fragments.
__global__ void gat_kernel(const bf16_t* __restrict__ hb, const bf16_t* __restrict__ WgT,
                           const bf16_t* __restrict__ WlinT, const float* __restrict__ al,
                           const float* __restrict__ ar, const float* __restrict__ blin,
                           const unsigned char* __restrict__ adj, bf16_t* __restrict__ zbf) {
  int g = blockIdx.x;            // g = bl*M + m
  int m = g & (MD - 1);
  int bl = g >> 2;
  int tid = threadIdx.x;
  int lane = tid & 31, wave = tid >> 5;

  const bf16_t* A    = hb + (size_t)bl * ND * DD;           // 16 x 768
  const bf16_t* Wg_m = WgT + (size_t)m * 64 * DD;           // [64][768] K-major
  const bf16_t* Wl_m = WlinT + (size_t)m * DD * 64;         // [768][64] K-major
  const float* al_m = al + m * 64;
  const float* ar_m = ar + m * 64;
  const unsigned char* adj_m = adj + (size_t)g * (ND * ND);

  __shared__ bf16_t hbS[16][DD];       // staged A tile
  __shared__ float  featS[16][64];
  __shared__ float  elS[2][16], erS[2][16];
  __shared__ float  alphaS[2][16][16];
  __shared__ bf16_t outbS[16][64];

  // ---- stage A tile into LDS ----
#if __has_builtin(__builtin_amdgcn_tensor_load_to_lds)
  if (wave == 0) {                         // TDM issues per wave; one wave is enough
    unsigned lds_off = (unsigned)(unsigned long long)(void*)&hbS[0][0];
    unsigned long long ga = (unsigned long long)(const void*)A;
    // D# group 0: count=1 | lds_addr | global_addr[56:0] | type=2
    u32x4 g0;
    g0[0] = 1u;
    g0[1] = lds_off;
    g0[2] = (unsigned)(ga & 0xffffffffu);
    g0[3] = (unsigned)((ga >> 32) & 0x01ffffffu) | (2u << 30);
    // D# group 1: data_size=1 (2B); 1-D tile: tensor_dim0 = tile_dim0 = 12288 elems
    i32x8 g1;
    g1[0] = (int)(1u << 16);            // data_size=2 bytes
    g1[1] = (int)(0x3000u << 16);       // tensor_dim0[15:0]=12288 (bits 63:48)
    g1[2] = (int)(1u << 16);            // tensor_dim0[31:16]=0, tensor_dim1[15:0]=1
    g1[3] = (int)(0x3000u << 16);       // tensor_dim1[31:16]=0, tile_dim0=12288
    g1[4] = 1;                          // tile_dim1=1, tile_dim2=0
    g1[5] = 12288;                      // tensor_dim0_stride[31:0]
    g1[6] = 0;                          // stride[47:32]=0, tensor_dim1_stride lo=0
    g1[7] = 0;
    i32x4 zz;
    zz[0] = 0; zz[1] = 0; zz[2] = 0; zz[3] = 0;
    i32x8 z8;
    z8[0] = 0; z8[1] = 0; z8[2] = 0; z8[3] = 0;
    z8[4] = 0; z8[5] = 0; z8[6] = 0; z8[7] = 0;
    __builtin_amdgcn_tensor_load_to_lds(g0, g1, zz, zz, z8, 0);
#if __has_builtin(__builtin_amdgcn_s_wait_tensorcnt)
    __builtin_amdgcn_s_wait_tensorcnt(0);
#else
    asm volatile("s_wait_tensorcnt 0x0" ::: "memory");
#endif
  }
  __syncthreads();
#else
  {
    const uint4* gs = (const uint4*)A;
    uint4* ls = (uint4*)&hbS[0][0];
    for (int i = tid; i < (ND * DD * 2) / 16; i += 128) ls[i] = gs[i];
  }
  __syncthreads();
#endif
  const bf16_t* Alds0 = &hbS[0][0];

  // feat = h @ Wg[m]  (16x768 * 768x64); wave w -> cols [16w,16w+16)
  {
    v8f acc;
#pragma unroll
    for (int i = 0; i < 8; ++i) acc[i] = 0.f;
    const bf16_t* Bcol = Wg_m + (size_t)(wave * 16) * DD;
    for (int k0 = 0; k0 < DD; k0 += 32)
      acc = wmma_bf16(load_frag_a(Alds0, DD, lane, k0),
                      load_frag_b(Bcol, DD, lane, k0), acc);
    int c = lane & 15, rb = (lane >> 4) << 3;
#pragma unroll
    for (int i = 0; i < 8; ++i) featS[rb + i][wave * 16 + c] = acc[i];
  }
  __syncthreads();

  // el/er: per (i,h) dot over O=32
  if (tid < 32) {
    int i = tid & 15, hh = tid >> 4;
    float se = 0.f, sr = 0.f;
    for (int o = 0; o < OD; ++o) {
      float f = featS[i][hh * OD + o];
      se += f * al_m[hh * OD + o];
      sr += f * ar_m[hh * OD + o];
    }
    elS[hh][i] = se;
    erS[hh][i] = sr;
  }
  __syncthreads();

  // masked softmax over j: e[i,j,h] = lrelu(er[i]+el[j], 0.2)
  if (tid < 32) {
    int i = tid & 15, hh = tid >> 4;
    float eri = erS[hh][i];
    float ev[16];
    float mx = -1e30f;
    for (int j = 0; j < ND; ++j) {
      float v = eri + elS[hh][j];
      v = v >= 0.f ? v : 0.2f * v;
      v = adj_m[i * ND + j] ? v : -1e9f;
      ev[j] = v;
      mx = fmaxf(mx, v);
    }
    float s = 0.f;
    for (int j = 0; j < ND; ++j) { ev[j] = expf(ev[j] - mx); s += ev[j]; }
    float inv = 1.f / s;
    for (int j = 0; j < ND; ++j)
      alphaS[hh][i][j] = adj_m[i * ND + j] ? ev[j] * inv : 0.f;
  }
  __syncthreads();

  // out = alpha @ feat, then ELU, store bf16 to LDS
#pragma unroll
  for (int e = 0; e < 8; ++e) {
    int idx = tid * 8 + e;
    int i = idx >> 6, col = idx & 63, hh = col >> 5;
    float s = 0.f;
    for (int j = 0; j < ND; ++j) s += alphaS[hh][i][j] * featS[j][col];
    s = s > 0.f ? s : expm1f(s);
    outbS[i][col] = (bf16_t)s;
  }
  __syncthreads();

  // z = out @ Wlin[m] + blin[m]; struct branch (m<M1) gets lrelu 0.1
  {
    int c = lane & 15, rb = (lane >> 4) << 3;
    const bf16_t* Alds = &outbS[0][0];
    for (int t = 0; t < 12; ++t) {
      int col0 = (wave * 12 + t) * 16;
      float bv = blin[m * DD + col0 + c];
      v8f acc;
#pragma unroll
      for (int i = 0; i < 8; ++i) acc[i] = bv;
      for (int k0 = 0; k0 < 64; k0 += 32)
        acc = wmma_bf16(load_frag_a(Alds, 64, lane, k0),
                        load_frag_b(Wl_m + (size_t)col0 * 64, 64, lane, k0), acc);
#pragma unroll
      for (int i = 0; i < 8; ++i) {
        float v = acc[i];
        if (m < M1D) v = v >= 0.f ? v : 0.1f * v;
        zbf[((size_t)g * ND + rb + i) * DD + col0 + c] = (bf16_t)v;
      }
    }
  }
}

// ---------------- semantic-attention scores: s = tanh(z@W1+b1)@w2 ----------------
// mode 0: rows from zbf m in {0,1}; mode 1: rows from zbf m in {2,3} and spool
__global__ void semscore_kernel(const bf16_t* __restrict__ zbf,
                                const bf16_t* __restrict__ spool,
                                const bf16_t* __restrict__ Wt, const float* __restrict__ b1,
                                const float* __restrict__ w2, float* __restrict__ sOut,
                                int mcount, int mode) {
  int t = blockIdx.x;
  int bl = t / mcount, mm = t - bl * mcount;
  const bf16_t* A;
  if (mode == 0)
    A = zbf + (size_t)(bl * MD + mm) * ND * DD;
  else
    A = (mm < 2) ? zbf + (size_t)(bl * MD + 2 + mm) * ND * DD
                 : spool + (size_t)bl * ND * DD;

  int lane = threadIdx.x & 31, wave = threadIdx.x >> 5;
  int c = lane & 15, rb = (lane >> 4) << 3;
  __shared__ float red[16];
  if (threadIdx.x < 16) red[threadIdx.x] = 0.f;
  __syncthreads();

  float rowpart[8];
#pragma unroll
  for (int i = 0; i < 8; ++i) rowpart[i] = 0.f;

  for (int tt = 0; tt < 2; ++tt) {
    int col0 = wave * 32 + tt * 16;
    float bv = b1[col0 + c];
    v8f acc;
#pragma unroll
    for (int i = 0; i < 8; ++i) acc[i] = bv;
    const bf16_t* Bcol = Wt + (size_t)col0 * DD;
    for (int k0 = 0; k0 < DD; k0 += 32)
      acc = wmma_bf16(load_frag_a(A, DD, lane, k0), load_frag_b(Bcol, DD, lane, k0), acc);
    float wv = w2[col0 + c];
#pragma unroll
    for (int i = 0; i < 8; ++i) rowpart[i] += tanhf(acc[i]) * wv;
  }
#pragma unroll
  for (int i = 0; i < 8; ++i) atomicAdd(&red[rb + i], rowpart[i]);
  __syncthreads();
  if (threadIdx.x < 16) sOut[(size_t)t * ND + threadIdx.x] = red[threadIdx.x];
}

// beta = softmax_m( mean_n s )
__global__ void beta_kernel(const float* __restrict__ s, float* __restrict__ beta, int mc) {
  int bl = blockIdx.x;
  if (threadIdx.x != 0) return;
  float mean[4];
  float mx = -1e30f;
  for (int mm = 0; mm < mc; ++mm) {
    float a = 0.f;
    for (int n = 0; n < ND; ++n) a += s[(size_t)(bl * mc + mm) * ND + n];
    mean[mm] = a * (1.f / ND);
    mx = fmaxf(mx, mean[mm]);
  }
  float se = 0.f;
  for (int mm = 0; mm < mc; ++mm) { mean[mm] = expf(mean[mm] - mx); se += mean[mm]; }
  float inv = 1.f / se;
  for (int mm = 0; mm < mc; ++mm) beta[bl * mc + mm] = mean[mm] * inv;
}

// s_pool[b,l,n,:] = sum_{m<2} beta_s[m] * struct[m]
__global__ void spool_kernel(const bf16_t* __restrict__ zbf, const float* __restrict__ beta,
                             bf16_t* __restrict__ spool) {
  size_t idx = (size_t)blockIdx.x * blockDim.x + threadIdx.x;
  int d = idx % DD;
  int n = (idx / DD) % ND;
  int bl = idx / ((size_t)DD * ND);
  float b0 = beta[bl * 2 + 0], b1 = beta[bl * 2 + 1];
  float v = b0 * (float)zbf[((size_t)(bl * MD + 0) * ND + n) * DD + d] +
            b1 * (float)zbf[((size_t)(bl * MD + 1) * ND + n) * DD + d];
  spool[idx] = (bf16_t)v;
}

// word_feat[b,l,:] = sum_m beta_w[m] * z2[b,l,n=0,m,:]
__global__ void wordfeat_kernel(const bf16_t* __restrict__ zbf,
                                const bf16_t* __restrict__ spool,
                                const float* __restrict__ betaw, bf16_t* __restrict__ wf) {
  int bl = blockIdx.x;
  float b0 = betaw[bl * 3 + 0], b1 = betaw[bl * 3 + 1], b2 = betaw[bl * 3 + 2];
#pragma unroll
  for (int j = 0; j < 3; ++j) {
    int d = threadIdx.x * 3 + j;
    float v = b0 * (float)zbf[((size_t)(bl * MD + 2) * ND) * DD + d] +
              b1 * (float)zbf[((size_t)(bl * MD + 3) * ND) * DD + d] +
              b2 * (float)spool[((size_t)bl * ND) * DD + d];
    wf[(size_t)bl * DD + d] = (bf16_t)v;
  }
}

// ---------------- generic WMMA GEMM: out = act(A@B + bias) ----------------
__global__ void gemm_bf16_kernel(const bf16_t* __restrict__ A, int lda,
                                 const bf16_t* __restrict__ Bt, int ldb,
                                 const float* __restrict__ bias,
                                 float* __restrict__ outF, int ldf,
                                 bf16_t* __restrict__ outB, int ldbo,
                                 int K, int act) {
  int lane = threadIdx.x & 31, wave = threadIdx.x >> 5;
  int row0 = blockIdx.x * 16;
  int col0 = blockIdx.y * 64 + wave * 16;
  const bf16_t* Arow = A + (size_t)row0 * lda;
  const bf16_t* Bcol = Bt + (size_t)col0 * ldb;
  int c = lane & 15, rb = (lane >> 4) << 3;
  float bv = bias ? bias[col0 + c] : 0.f;
  v8f acc;
#pragma unroll
  for (int i = 0; i < 8; ++i) acc[i] = bv;
  for (int k0 = 0; k0 < K; k0 += 32) {
    if (k0 + 128 < K)
      __builtin_prefetch(Bcol + (size_t)c * ldb + k0 + 128, 0, 0);
    v16bf a = load_frag_a(Arow, lda, lane, k0);
    v16bf b = load_frag_b(Bcol, ldb, lane, k0);
    acc = wmma_bf16(a, b, acc);
  }
#pragma unroll
  for (int i = 0; i < 8; ++i) {
    float v = apply_act(acc[i], act);
    size_t r = (size_t)(row0 + rb + i);
    if (outF) outF[r * ldf + col0 + c] = v;
    if (outB) outB[r * ldbo + col0 + c] = (bf16_t)v;
  }
}

// bert_feats -> first half of xcat (bf16, row stride 1536)
__global__ void bert_to_xcat(const float* __restrict__ bert, bf16_t* __restrict__ xcat) {
  size_t idx = (size_t)blockIdx.x * blockDim.x + threadIdx.x;
  int r = idx / DD, d = idx % DD;
  xcat[(size_t)r * (2 * DD) + d] = (bf16_t)bert[idx];
}

// ---------------- self-attention over L=6, per (b, head) ----------------
__global__ void attn_kernel(const float* __restrict__ qf, const float* __restrict__ kf,
                            const float* __restrict__ vf, bf16_t* __restrict__ ab) {
  int g = blockIdx.x;
  int h = g % NHD, b = g / NHD;
  int t = threadIdx.x;  // 64 threads
  __shared__ float sc[LD][LD];
  size_t base = (size_t)b * LD * DD + (size_t)h * DHD;
  if (t < LD * LD) {
    int l = t / LD, mm = t - l * LD;
    float s = 0.f;
    for (int d = 0; d < DHD; ++d)
      s += qf[base + (size_t)l * DD + d] * kf[base + (size_t)mm * DD + d];
    sc[l][mm] = s * 0.125f;  // 1/sqrt(64)
  }
  __syncthreads();
  if (t < LD) {
    float mx = -1e30f;
    for (int mm = 0; mm < LD; ++mm) mx = fmaxf(mx, sc[t][mm]);
    float s = 0.f;
    for (int mm = 0; mm < LD; ++mm) { sc[t][mm] = expf(sc[t][mm] - mx); s += sc[t][mm]; }
    float inv = 1.f / s;
    for (int mm = 0; mm < LD; ++mm) sc[t][mm] *= inv;
  }
  __syncthreads();
  for (int l = 0; l < LD; ++l) {
    float s = 0.f;
    for (int mm = 0; mm < LD; ++mm) s += sc[l][mm] * vf[base + (size_t)mm * DD + t];
    ab[base + (size_t)l * DD + t] = (bf16_t)s;
  }
}

// ---------------- residual + layernorm ----------------
__global__ void ln_kernel(const float* __restrict__ x, const float* __restrict__ r,
                          const float* __restrict__ g, const float* __restrict__ be,
                          float* __restrict__ outF, bf16_t* __restrict__ outB) {
  int row = blockIdx.x, tid = threadIdx.x;  // 256 threads, D=768
  __shared__ float red[256];
  float v[3];
  float s = 0.f;
#pragma unroll
  for (int j = 0; j < 3; ++j) {
    int d = tid * 3 + j;
    float val = x[(size_t)row * DD + d] + r[(size_t)row * DD + d];
    v[j] = val;
    s += val;
  }
  red[tid] = s;
  __syncthreads();
  for (int st = 128; st > 0; st >>= 1) {
    if (tid < st) red[tid] += red[tid + st];
    __syncthreads();
  }
  float mu = red[0] * (1.f / DD);
  __syncthreads();
  s = 0.f;
#pragma unroll
  for (int j = 0; j < 3; ++j) { float d2 = v[j] - mu; s += d2 * d2; }
  red[tid] = s;
  __syncthreads();
  for (int st = 128; st > 0; st >>= 1) {
    if (tid < st) red[tid] += red[tid + st];
    __syncthreads();
  }
  float rstd = rsqrtf(red[0] * (1.f / DD) + 1e-12f);
#pragma unroll
  for (int j = 0; j < 3; ++j) {
    int d = tid * 3 + j;
    float o = (v[j] - mu) * rstd * g[d] + be[d];
    outF[(size_t)row * DD + d] = o;
    if (outB) outB[(size_t)row * DD + d] = (bf16_t)o;
  }
}

// gather l==0 rows as bf16
__global__ void pool_gather(const float* __restrict__ x2f, bf16_t* __restrict__ pin) {
  size_t idx = (size_t)blockIdx.x * blockDim.x + threadIdx.x;
  int b = idx / DD, d = idx % DD;
  pin[idx] = (bf16_t)x2f[((size_t)b * LD) * DD + d];
}

// logits = pooled @ Wm + bm   (256 x 10)
__global__ void logits_kernel(const float* __restrict__ pooled, const float* __restrict__ Wm,
                              const float* __restrict__ bm, float* __restrict__ out) {
  int b = blockIdx.x, c = threadIdx.x;
  if (c >= NLB) return;
  float s = bm[c];
  for (int d = 0; d < DD; ++d) s += pooled[(size_t)b * DD + d] * Wm[(size_t)d * NLB + c];
  out[(size_t)b * NLB + c] = s;
}

// =====================================================================
extern "C" void kernel_launch(void* const* d_in, const int* in_sizes, int n_in,
                              void* d_out, int out_size, void* d_ws, size_t ws_size,
                              hipStream_t stream) {
  const int*           word_ids = (const int*)d_in[0];
  const int*           pos_ids  = (const int*)d_in[1];
  const unsigned char* adj      = (const unsigned char*)d_in[2];
  const float* bert  = (const float*)d_in[3];
  const float* we    = (const float*)d_in[4];
  const float* pe    = (const float*)d_in[5];
  const float* Wg    = (const float*)d_in[6];
  const float* al    = (const float*)d_in[7];
  const float* ar    = (const float*)d_in[8];
  const float* Wlin  = (const float*)d_in[9];
  const float* blin  = (const float*)d_in[10];
  const float* Ws1   = (const float*)d_in[11];
  const float* bs1   = (const float*)d_in[12];
  const float* ws2   = (const float*)d_in[13];
  const float* Ww1   = (const float*)d_in[14];
  const float* bw1   = (const float*)d_in[15];
  const float* ww2   = (const float*)d_in[16];
  const float* W1    = (const float*)d_in[17];
  const float* b1    = (const float*)d_in[18];
  const float* Wcat  = (const float*)d_in[19];
  const float* bcat  = (const float*)d_in[20];
  const float* Wq    = (const float*)d_in[21];
  const float* bq    = (const float*)d_in[22];
  const float* Wk    = (const float*)d_in[23];
  const float* bk    = (const float*)d_in[24];
  const float* Wv    = (const float*)d_in[25];
  const float* bv    = (const float*)d_in[26];
  const float* Wo    = (const float*)d_in[27];
  const float* bo    = (const float*)d_in[28];
  const float* g1    = (const float*)d_in[29];
  const float* be1   = (const float*)d_in[30];
  const float* Wf1   = (const float*)d_in[31];
  const float* bf1   = (const float*)d_in[32];
  const float* Wf2   = (const float*)d_in[33];
  const float* bf2   = (const float*)d_in[34];
  const float* g2    = (const float*)d_in[35];
  const float* be2   = (const float*)d_in[36];
  const float* Wp    = (const float*)d_in[37];
  const float* bp    = (const float*)d_in[38];
  const float* Wm    = (const float*)d_in[39];
  const float* bm    = (const float*)d_in[40];
  float* out = (float*)d_out;

  // ---- carve workspace ----
  char* ws = (char*)d_ws;
  size_t off = 0;
  auto carve = [&](size_t bytes) -> void* {
    off = (off + 255) & ~(size_t)255;
    void* p = ws + off;
    off += bytes;
    return p;
  };
  bf16_t* WgT   = (bf16_t*)carve((size_t)MD * 64 * DD * 2);
  bf16_t* WlinT = (bf16_t*)carve((size_t)MD * DD * 64 * 2);
  bf16_t* Ws1T  = (bf16_t*)carve((size_t)HIDD * DD * 2);
  bf16_t* Ww1T  = (bf16_t*)carve((size_t)HIDD * DD * 2);
  bf16_t* W1T   = (bf16_t*)carve((size_t)DD * DD * 2);
  bf16_t* WcatT = (bf16_t*)carve((size_t)2 * DD * DD * 2);
  bf16_t* WqT   = (bf16_t*)carve((size_t)DD * DD * 2);
  bf16_t* WkT   = (bf16_t*)carve((size_t)DD * DD * 2);
  bf16_t* WvT   = (bf16_t*)carve((size_t)DD * DD * 2);
  bf16_t* WoT   = (bf16_t*)carve((size_t)DD * DD * 2);
  bf16_t* Wf1T  = (bf16_t*)carve((size_t)FFD * DD * 2);
  bf16_t* Wf2T  = (bf16_t*)carve((size_t)DD * FFD * 2);
  bf16_t* WpT   = (bf16_t*)carve((size_t)DD * DD * 2);
  bf16_t* hb    = (bf16_t*)carve((size_t)TOK * DD * 2);
  bf16_t* zbf   = (bf16_t*)carve((size_t)BLD * MD * ND * DD * 2);
  float*  s_s   = (float*)carve((size_t)BLD * 2 * ND * 4);
  float*  s_w   = (float*)carve((size_t)BLD * 3 * ND * 4);
  float*  bet_s = (float*)carve((size_t)BLD * 2 * 4);
  float*  bet_w = (float*)carve((size_t)BLD * 3 * 4);
  bf16_t* spool = (bf16_t*)carve((size_t)BLD * ND * DD * 2);
  bf16_t* wfb   = (bf16_t*)carve((size_t)BLD * DD * 2);
  bf16_t* xcat  = (bf16_t*)carve((size_t)BLD * 2 * DD * 2);
  float*  x0f   = (float*)carve((size_t)BLD * DD * 4);
  bf16_t* x0b   = (bf16_t*)carve((size_t)BLD * DD * 2);
  float*  qf    = (float*)carve((size_t)BLD * DD * 4);
  float*  kf    = (float*)carve((size_t)BLD * DD * 4);
  float*  vf    = (float*)carve((size_t)BLD * DD * 4);
  bf16_t* ab    = (bf16_t*)carve((size_t)BLD * DD * 2);
  float*  woF   = (float*)carve((size_t)BLD * DD * 4);
  float*  x1f   = (float*)carve((size_t)BLD * DD * 4);
  bf16_t* x1b   = (bf16_t*)carve((size_t)BLD * DD * 2);
  bf16_t* f1b   = (bf16_t*)carve((size_t)BLD * FFD * 2);
  float*  f2f   = (float*)carve((size_t)BLD * DD * 4);
  float*  x2f   = (float*)carve((size_t)BLD * DD * 4);
  bf16_t* pin   = (bf16_t*)carve((size_t)BD * DD * 2);
  float*  poolf = (float*)carve((size_t)BD * DD * 4);
  (void)ws_size; (void)n_in; (void)in_sizes; (void)out_size;

  // ---- weight transposes (fp32 -> bf16 K-major) ----
  auto tr = [&](const float* W, bf16_t* Wt, int batch, int K, int Nc) {
    int total = batch * K * Nc;
    transpose_to_bf16<<<(total + 255) / 256, 256, 0, stream>>>(W, Wt, K, Nc, total);
  };
  tr(Wg,   WgT,   MD, DD, 64);
  tr(Wlin, WlinT, MD, 64, DD);
  tr(Ws1,  Ws1T,  1, DD, HIDD);
  tr(Ww1,  Ww1T,  1, DD, HIDD);
  tr(W1,   W1T,   1, DD, DD);
  tr(Wcat, WcatT, 1, 2 * DD, DD);
  tr(Wq,   WqT,   1, DD, DD);
  tr(Wk,   WkT,   1, DD, DD);
  tr(Wv,   WvT,   1, DD, DD);
  tr(Wo,   WoT,   1, DD, DD);
  tr(Wf1,  Wf1T,  1, DD, FFD);
  tr(Wf2,  Wf2T,  1, FFD, DD);
  tr(Wp,   WpT,   1, DD, DD);

  // ---- embeddings ----
  embed_kernel<<<TOK, 256, 0, stream>>>(word_ids, pos_ids, we, pe, hb);

  // ---- GAT per (b,l,m) ----
  gat_kernel<<<BLD * MD, 128, 0, stream>>>(hb, WgT, WlinT, al, ar, blin, adj, zbf);

  // ---- structural semantic attention ----
  semscore_kernel<<<BLD * 2, 128, 0, stream>>>(zbf, spool, Ws1T, bs1, ws2, s_s, 2, 0);
  beta_kernel<<<BLD, 32, 0, stream>>>(s_s, bet_s, 2);
  spool_kernel<<<(BLD * ND * DD) / 256, 256, 0, stream>>>(zbf, bet_s, spool);

  // ---- word-level semantic attention ----
  semscore_kernel<<<BLD * 3, 128, 0, stream>>>(zbf, spool, Ww1T, bw1, ww2, s_w, 3, 1);
  beta_kernel<<<BLD, 32, 0, stream>>>(s_w, bet_w, 3);
  wordfeat_kernel<<<BLD, 256, 0, stream>>>(zbf, spool, bet_w, wfb);

  dim3 blk(128);
  // x1 = lrelu(wf @ W1 + b1) -> second half of xcat
  gemm_bf16_kernel<<<dim3(BLD / 16, DD / 64), blk, 0, stream>>>(
      wfb, DD, W1T, DD, b1, nullptr, 0, xcat + DD, 2 * DD, DD, 1);
  bert_to_xcat<<<(BLD * DD) / 256, 256, 0, stream>>>(bert, xcat);

  // x = lrelu(xcat @ Wcat + bcat)
  gemm_bf16_kernel<<<dim3(BLD / 16, DD / 64), blk, 0, stream>>>(
      xcat, 2 * DD, WcatT, 2 * DD, bcat, x0f, DD, x0b, DD, 2 * DD, 1);

  // q,k,v
  gemm_bf16_kernel<<<dim3(BLD / 16, DD / 64), blk, 0, stream>>>(
      x0b, DD, WqT, DD, bq, qf, DD, nullptr, 0, DD, 0);
  gemm_bf16_kernel<<<dim3(BLD / 16, DD / 64), blk, 0, stream>>>(
      x0b, DD, WkT, DD, bk, kf, DD, nullptr, 0, DD, 0);
  gemm_bf16_kernel<<<dim3(BLD / 16, DD / 64), blk, 0, stream>>>(
      x0b, DD, WvT, DD, bv, vf, DD, nullptr, 0, DD, 0);

  // attention over L=6
  attn_kernel<<<BD * NHD, 64, 0, stream>>>(qf, kf, vf, ab);

  // Wo projection + residual LN1
  gemm_bf16_kernel<<<dim3(BLD / 16, DD / 64), blk, 0, stream>>>(
      ab, DD, WoT, DD, bo, woF, DD, nullptr, 0, DD, 0);
  ln_kernel<<<BLD, 256, 0, stream>>>(woF, x0f, g1, be1, x1f, x1b);

  // FFN
  gemm_bf16_kernel<<<dim3(BLD / 16, FFD / 64), blk, 0, stream>>>(
      x1b, DD, Wf1T, DD, bf1, nullptr, 0, f1b, FFD, DD, 2);
  gemm_bf16_kernel<<<dim3(BLD / 16, DD / 64), blk, 0, stream>>>(
      f1b, FFD, Wf2T, FFD, bf2, f2f, DD, nullptr, 0, FFD, 0);
  ln_kernel<<<BLD, 256, 0, stream>>>(f2f, x1f, g2, be2, x2f, nullptr);

  // pooled = tanh(x2[:,0] @ Wp + bp); logits = pooled @ Wm + bm
  pool_gather<<<(BD * DD) / 256, 256, 0, stream>>>(x2f, pin);
  gemm_bf16_kernel<<<dim3(BD / 16, DD / 64), blk, 0, stream>>>(
      pin, DD, WpT, DD, bp, poolf, DD, nullptr, 0, DD, 3);
  logits_kernel<<<BD, 32, 0, stream>>>(poolf, Wm, bm, out);
}
